// GIN_AttProj_76888504533071
// MI455X (gfx1250) — matrix operand order repaired and checked by reading, not compiled, not run on hardware
//
#include <hip/hip_runtime.h>
#include <hip/hip_bf16.h>
#include <stddef.h>

typedef __bf16  v16bf __attribute__((ext_vector_type(16)));
typedef float   v8f   __attribute__((ext_vector_type(8)));
typedef float   v4f   __attribute__((ext_vector_type(4)));

#define K_DIM   128
#define GH       64
#define OUT_DIM 128

__global__ void zero_f32_kernel(float* __restrict__ p, int n) {
    int i = blockIdx.x * blockDim.x + threadIdx.x;
    if (i < n) p[i] = 0.0f;
}

__device__ __forceinline__ void cvt8(const v4f& a, const v4f& b, v16bf& dst, int base) {
    dst[base + 0] = (__bf16)a[0]; dst[base + 1] = (__bf16)a[1];
    dst[base + 2] = (__bf16)a[2]; dst[base + 3] = (__bf16)a[3];
    dst[base + 4] = (__bf16)b[0]; dst[base + 5] = (__bf16)b[1];
    dst[base + 6] = (__bf16)b[2]; dst[base + 7] = (__bf16)b[3];
}

// One wave32 processes a 16-node tile:
//   A (16x128 bf16, registers, NT-streamed from global)  x  B-fragments (LDS-staged)
//   -> v_wmma_f32_16x16x32_bf16 with f32 accumulation.
// gate = sigmoid(relu(h@Wg1+b1)@Wg2+b2) via half-wave shuffle reduction,
// gated proj rows run-compressed by sorted graph id -> global_atomic_add_f32.
__global__ void __launch_bounds__(256)
gin_attproj_kernel(const float* __restrict__ h,
                   const int*   __restrict__ batch,
                   const float* __restrict__ Wg1,   // [128,64]
                   const float* __restrict__ bg1,   // [64]
                   const float* __restrict__ Wg2,   // [64,1]
                   const float* __restrict__ bg2,   // [1]
                   const float* __restrict__ Wp,    // [128,128]
                   const float* __restrict__ bp,    // [128]
                   float* __restrict__ out,         // [B,128]
                   int nTiles)
{
    // ---- LDS: B-matrix fragments in exact per-lane WMMA order ----
    // fragment f=(t*4+c): 32 lanes x 16 bf16 contiguous (32B per lane)
    __shared__ __align__(32) __bf16 sBp[32 * 512];  // Wp : 8 n-tiles x 4 k-chunks
    __shared__ __align__(32) __bf16 sBg[16 * 512];  // Wg1: 4 n-tiles x 4 k-chunks
    __shared__ float sB1[GH];
    __shared__ float sW2[GH];
    __shared__ float sBP[OUT_DIM];

    const int tid = threadIdx.x;

    for (int idx = tid; idx < 32 * 512; idx += blockDim.x) {
        int f = idx >> 9, rem = idx & 511;
        int lane = rem >> 4, e = rem & 15;
        int t = f >> 2, c = f & 3;
        int k = c * 32 + ((lane & 16) ? 8 : 0) + (e & 7) + ((e & 8) ? 16 : 0);   // B layout: lanes>=16 hold K=8..15,24..31
        int n = t * 16 + (lane & 15);
        sBp[idx] = (__bf16)Wp[k * OUT_DIM + n];
    }
    for (int idx = tid; idx < 16 * 512; idx += blockDim.x) {
        int f = idx >> 9, rem = idx & 511;
        int lane = rem >> 4, e = rem & 15;
        int t = f >> 2, c = f & 3;
        int k = c * 32 + ((lane & 16) ? 8 : 0) + (e & 7) + ((e & 8) ? 16 : 0);
        int n = t * 16 + (lane & 15);
        sBg[idx] = (__bf16)Wg1[k * GH + n];
    }
    if (tid < GH)      { sB1[tid] = bg1[tid]; sW2[tid] = Wg2[tid]; }
    if (tid < OUT_DIM) { sBP[tid] = bp[tid]; }
    __syncthreads();

    const float bg2v   = bg2[0];
    const int   lane   = tid & 31;
    const int   wave   = tid >> 5;
    const int   laneM  = lane & 15;
    const bool  hiHalf = lane >= 16;           // owns C rows 8..15
    const int   wpb    = blockDim.x >> 5;

    for (int tile = blockIdx.x * wpb + wave; tile < nTiles; tile += gridDim.x * wpb) {
        const int rowBase = tile * 16;

        // ---- A fragments: 4 K-chunks, ISA 16-bit A layout ----
        // 512MB one-shot stream vs 192MB L2 -> non-temporal so the stream does
        // not evict the atomic output region / weight tables from L2.
        v16bf A[4];
        const float* hrow = h + (size_t)(rowBase + laneM) * K_DIM;
        const int kofs = hiHalf ? 8 : 0;
#pragma unroll
        for (int c = 0; c < 4; ++c) {
            v4f f0 = __builtin_nontemporal_load((const v4f*)(hrow + c * 32 + kofs));
            v4f f1 = __builtin_nontemporal_load((const v4f*)(hrow + c * 32 + kofs + 4));
            v4f f2 = __builtin_nontemporal_load((const v4f*)(hrow + c * 32 + kofs + 16));
            v4f f3 = __builtin_nontemporal_load((const v4f*)(hrow + c * 32 + kofs + 20));
            cvt8(f0, f1, A[c], 0);
            cvt8(f2, f3, A[c], 8);
        }

        // ---- gate GEMM: hidden = h @ Wg1  (4 C-tiles of 16x16) ----
        v8f Cg[4];
#pragma unroll
        for (int t = 0; t < 4; ++t) {
            v8f acc = {};
#pragma unroll
            for (int c = 0; c < 4; ++c) {
                v16bf B = *(const v16bf*)&sBg[(size_t)(t * 4 + c) * 512 + lane * 16];
                acc = __builtin_amdgcn_wmma_f32_16x16x32_bf16(
                          false, A[c], false, B, (short)0, acc, false, false);
            }
            Cg[t] = acc;
        }

        // ---- gate scalar: relu(+b1) . Wg2, reduce across the 16-lane row group ----
        float gate[8];
#pragma unroll
        for (int r = 0; r < 8; ++r) {
            float p = 0.0f;
#pragma unroll
            for (int t = 0; t < 4; ++t) {
                int col = t * 16 + laneM;
                float v = fmaxf(Cg[t][r] + sB1[col], 0.0f);
                p += v * sW2[col];
            }
            p += __shfl_xor(p, 1, 32);
            p += __shfl_xor(p, 2, 32);
            p += __shfl_xor(p, 4, 32);
            p += __shfl_xor(p, 8, 32);
            gate[r] = 1.0f / (1.0f + __expf(-(p + bg2v)));
        }

        // ---- graph ids for my 8 rows (batch is sorted; 4MB one-shot -> NT) ----
        int gids[8];
        const int rb = rowBase + (hiHalf ? 8 : 0);
#pragma unroll
        for (int r = 0; r < 8; ++r)
            gids[r] = __builtin_nontemporal_load(batch + rb + r);

        // ---- proj GEMM + gated run-compressed scatter-add ----
#pragma unroll
        for (int t = 0; t < 8; ++t) {
            v8f acc = {};
#pragma unroll
            for (int c = 0; c < 4; ++c) {
                v16bf B = *(const v16bf*)&sBp[(size_t)(t * 4 + c) * 512 + lane * 16];
                acc = __builtin_amdgcn_wmma_f32_16x16x32_bf16(
                          false, A[c], false, B, (short)0, acc, false, false);
            }
            const int col = t * 16 + laneM;
            const float bpv = sBP[col];
            float run = 0.0f;
            int   prev = gids[0];
#pragma unroll
            for (int r = 0; r < 8; ++r) {
                int g = gids[r];
                if (g != prev) {
                    atomicAdd(out + (size_t)prev * OUT_DIM + col, run);
                    run = 0.0f;
                    prev = g;
                }
                run += gate[r] * (acc[r] + bpv);
            }
            atomicAdd(out + (size_t)prev * OUT_DIM + col, run);
        }
    }
}

extern "C" void kernel_launch(void* const* d_in, const int* in_sizes, int n_in,
                              void* d_out, int out_size, void* d_ws, size_t ws_size,
                              hipStream_t stream) {
    const float* h     = (const float*)d_in[0];
    const int*   batch = (const int*)  d_in[1];
    const float* Wg1   = (const float*)d_in[2];
    const float* bg1   = (const float*)d_in[3];
    const float* Wg2   = (const float*)d_in[4];
    const float* bg2   = (const float*)d_in[5];
    const float* Wp    = (const float*)d_in[6];
    const float* bp    = (const float*)d_in[7];
    float* out = (float*)d_out;

    const int nNodes = in_sizes[1];     // batch element count == NUM_NODES
    const int nTiles = nNodes / 16;     // 1,000,000 / 16 = 62,500

    zero_f32_kernel<<<(out_size + 255) / 256, 256, 0, stream>>>(out, out_size);

    // 256 threads = 8 wave32 per block, each wave owns a 16-node tile (grid-stride)
    const int blocks = 1280;
    gin_attproj_kernel<<<blocks, 256, 0, stream>>>(h, batch, Wg1, bg1, Wg2, bg2,
                                                   Wp, bp, out, nTiles);
}
